// LutLinear_24859270709249
// MI455X (gfx1250) — matrix operand order repaired and checked by reading, not compiled, not run on hardware
//
#include <hip/hip_runtime.h>
#include <hip/hip_bf16.h>

// ---------------------------------------------------------------------------
// Problem constants (from the reference)
// ---------------------------------------------------------------------------
#define O_DIM   4096
#define I_DIM   4096
#define S_DIM   512
#define GS      128
#define NGROUP  (I_DIM / GS)   // 32
#define R_DIM   64
#define MAXQ    15.0f
#define QEPS    1e-5f

typedef _Float16 v16h __attribute__((ext_vector_type(16)));
typedef _Float16 v8h  __attribute__((ext_vector_type(8)));
typedef float    v8f  __attribute__((ext_vector_type(8)));

// ---------------------------------------------------------------------------
// CDNA5 async load: HBM -> LDS directly (no VGPR round trip), ASYNCcnt-tracked
// LDS generic pointers carry the LDS byte offset in their low 32 bits.
// ---------------------------------------------------------------------------
__device__ __forceinline__ void async_b128(const void* lds_ptr, const void* gptr) {
    uint32_t loff = (uint32_t)(uintptr_t)lds_ptr;
    uint64_t ga   = (uint64_t)(uintptr_t)gptr;
    asm volatile("global_load_async_to_lds_b128 %0, %1, off"
                 :: "v"(loff), "v"(ga) : "memory");
}
__device__ __forceinline__ void wait_async3() {
    asm volatile("s_wait_asynccnt 0x3" ::: "memory");
}
__device__ __forceinline__ void wait_async0() {
    asm volatile("s_wait_asynccnt 0x0" ::: "memory");
}

// ---------------------------------------------------------------------------
// Kernel 1: convert x (f32) -> f16, 8 elements / thread, fully coalesced
// ---------------------------------------------------------------------------
__global__ __launch_bounds__(256) void cvt_x_kernel(const float* __restrict__ x,
                                                    _Float16* __restrict__ xh) {
    const int idx = (blockIdx.x * 256 + threadIdx.x) * 8;
    const float4 a = *(const float4*)(x + idx);
    const float4 b = *(const float4*)(x + idx + 4);
    v8h h;
    h[0] = (_Float16)a.x; h[1] = (_Float16)a.y; h[2] = (_Float16)a.z; h[3] = (_Float16)a.w;
    h[4] = (_Float16)b.x; h[5] = (_Float16)b.y; h[6] = (_Float16)b.z; h[7] = (_Float16)b.w;
    *(v8h*)(xh + idx) = h;
}

// ---------------------------------------------------------------------------
// Kernel 2: fake-quantize weight + low-rank correction -> Wq[O][I] (f16)
// ---------------------------------------------------------------------------
__global__ __launch_bounds__(128) void quant_kernel(const float* __restrict__ weight,
                                                    const float* __restrict__ factor,
                                                    const float* __restrict__ col_scale,
                                                    const float* __restrict__ col_bias,
                                                    const float* __restrict__ Amat,
                                                    const float* __restrict__ Bmat,
                                                    _Float16* __restrict__ wq) {
    __shared__ float red[GS];
    const int g = blockIdx.x;          // 0..31
    const int o = blockIdx.y;          // 0..4095
    const int t = threadIdx.x;         // 0..127
    const int i = g * GS + t;

    const float cs = fmaxf(col_scale[i], QEPS);
    const float cb = col_bias[i];
    const float wn = (weight[(size_t)o * I_DIM + i] - cb) / cs;   // normalized

    red[t] = wn;
    __syncthreads();
    for (int s = GS / 2; s > 0; s >>= 1) {
        if (t < s) red[t] = fmaxf(red[t], red[t + s]);
        __syncthreads();
    }
    const float gmax = red[0];

    const float f   = factor[o * NGROUP + g];
    const float sig = 1.0f / (1.0f + __expf(-f));
    const float rs  = fmaxf(gmax * sig, QEPS);

    float q = rintf(wn / rs * MAXQ);            // round half to even
    q = fminf(fmaxf(q, 0.0f), MAXQ) * (1.0f / MAXQ);
    float wdq = q * rs * cs + cb;

    float lr = 0.0f;
#pragma unroll 8
    for (int r = 0; r < R_DIM; ++r)
        lr += Amat[o * R_DIM + r] * Bmat[(size_t)r * I_DIM + i];

    wq[(size_t)o * I_DIM + i] = (_Float16)(wdq + lr);
}

// ---------------------------------------------------------------------------
// Kernel 3: transpose Wq[O][I] -> WqT[I][O] (f16), 64x64 tiles through LDS
// ---------------------------------------------------------------------------
__global__ __launch_bounds__(256) void trans_kernel(const _Float16* __restrict__ wq,
                                                    _Float16* __restrict__ wqt) {
    __shared__ __align__(16) _Float16 T[64][72];
    const int o0 = blockIdx.x * 64;
    const int i0 = blockIdx.y * 64;
    const int t  = threadIdx.x;
    const int r0 = t >> 3;          // 0..31
    const int c  = (t & 7) * 8;     // 0,8,..,56

#pragma unroll
    for (int rr = r0; rr < 64; rr += 32) {
        *(uint4*)&T[rr][c] = *(const uint4*)&wq[(size_t)(o0 + rr) * I_DIM + i0 + c];
    }
    __syncthreads();
#pragma unroll
    for (int rr = r0; rr < 64; rr += 32) {     // rr = i-local
        v8h h;
#pragma unroll
        for (int j = 0; j < 8; ++j) h[j] = T[c + j][rr];
        *(v8h*)&wqt[(size_t)(i0 + rr) * O_DIM + o0 + c] = h;
    }
}

// ---------------------------------------------------------------------------
// Kernel 4: WMMA GEMM  out[s][o] = sum_i xh[s][i] * wqT[i][o] + bias[o]
// 256 threads (8 waves) per block compute a 64(M) x 128(N) tile.
// Waves arranged 2(M) x 4(N); each wave owns a 32x32 output block =
// 2x2 grid of 16x16 WMMA tiles -> 4 v_wmma per K-step from 2 A + 2 B frags.
// Double-buffered LDS fed by global_load_async_to_lds_b128 (ASYNCcnt).
// ---------------------------------------------------------------------------
__global__ __launch_bounds__(256) void gemm_kernel(const _Float16* __restrict__ xh,
                                                   const _Float16* __restrict__ wqt,
                                                   const float* __restrict__ bias,
                                                   float* __restrict__ out) {
    // padded row strides (bytes): 48*2=96, 136*2=272 — both multiples of 16
    __shared__ __align__(16) _Float16 Als[2][64][48];    // A tile: [m][k]
    __shared__ __align__(16) _Float16 Bls[2][32][136];   // B tile: [k][n]

    const int t    = threadIdx.x;
    const int lane = t & 31;
    const int wv   = t >> 5;               // 0..7
    const int m0   = blockIdx.x * 64;      // S tile base
    const int n0   = blockIdx.y * 128;     // O tile base
    const int mw   = (wv & 1) * 32;        // wave M offset in tile
    const int nw   = (wv >> 1) * 32;       // wave N offset in tile

    // staging assignments (3 async b128 loads per thread per K-step)
    const int am  = t >> 2,  ac = (t & 3) * 8;     // A: 64 rows x 4 chunks
    const int bk0 = t >> 4,  bc = (t & 15) * 8;    // B: rows bk0, bk0+16

    // A-fragment addressing (16x32 f16): lane = M | K-group, halves = K
    const int fm  = lane & 15;
    const int grp = lane >> 4;             // 0: K 0-7/16-23, 1: K 8-15/24-31

    v8f c00 = {}, c01 = {}, c10 = {}, c11 = {};

    // ---- prologue: stage K-step 0 into buffer 0
    async_b128(&Als[0][am][ac],        &xh[(size_t)(m0 + am) * I_DIM + ac]);
    async_b128(&Bls[0][bk0][bc],       &wqt[(size_t)bk0 * O_DIM + n0 + bc]);
    async_b128(&Bls[0][bk0 + 16][bc],  &wqt[(size_t)(bk0 + 16) * O_DIM + n0 + bc]);

    for (int kk = 0; kk < I_DIM; kk += 32) {
        const int cur = (kk >> 5) & 1;
        const int nxt = cur ^ 1;
        const int kn  = kk + 32;

        if (kn < I_DIM) {
            // prefetch next K-step into the other buffer, then wait only for
            // the previous 3 loads (ours for `cur`) to land in LDS
            async_b128(&Als[nxt][am][ac],       &xh[(size_t)(m0 + am) * I_DIM + kn + ac]);
            async_b128(&Bls[nxt][bk0][bc],      &wqt[(size_t)(kn + bk0) * O_DIM + n0 + bc]);
            async_b128(&Bls[nxt][bk0 + 16][bc], &wqt[(size_t)(kn + bk0 + 16) * O_DIM + n0 + bc]);
            wait_async3();
        } else {
            wait_async0();
        }
        __syncthreads();   // all waves' `cur` tile now visible in LDS

        // ---- A fragments (rows mw.. and mw+16..)
        const v8h a0lo = *(const v8h*)&Als[cur][mw + fm][grp * 8];
        const v8h a0hi = *(const v8h*)&Als[cur][mw + fm][16 + grp * 8];
        const v8h a1lo = *(const v8h*)&Als[cur][mw + 16 + fm][grp * 8];
        const v8h a1hi = *(const v8h*)&Als[cur][mw + 16 + fm][16 + grp * 8];
        v16h a0, a1;
#pragma unroll
        for (int j = 0; j < 8; ++j) {
            a0[j] = a0lo[j]; a0[j + 8] = a0hi[j];
            a1[j] = a1lo[j]; a1[j + 8] = a1hi[j];
        }

        // ---- B fragments (cols nw.. and nw+16..): lane = N | K-group, halves = K
        v16h b0, b1;
#pragma unroll
        for (int j = 0; j < 8; ++j) {
            b0[j] = Bls[cur][grp * 8 + j][nw + fm]; b0[j + 8] = Bls[cur][16 + grp * 8 + j][nw + fm];
            b1[j] = Bls[cur][grp * 8 + j][nw + 16 + fm]; b1[j + 8] = Bls[cur][16 + grp * 8 + j][nw + 16 + fm];
        }

        // ---- 4 WMMAs per wave per K-step (f32 accumulate)
        c00 = __builtin_amdgcn_wmma_f32_16x16x32_f16(false, a0, false, b0,
                                                     (short)0, c00, false, false);
        c01 = __builtin_amdgcn_wmma_f32_16x16x32_f16(false, a0, false, b1,
                                                     (short)0, c01, false, false);
        c10 = __builtin_amdgcn_wmma_f32_16x16x32_f16(false, a1, false, b0,
                                                     (short)0, c10, false, false);
        c11 = __builtin_amdgcn_wmma_f32_16x16x32_f16(false, a1, false, b1,
                                                     (short)0, c11, false, false);

        __syncthreads();   // everyone done reading `cur` before it is refilled
    }

    // ---- epilogue: C/D layout: lane 0-15 -> N=lane, M=vgpr; lane 16-31 -> M=vgpr+8
    const int nl    = lane & 15;
    const int mbase = (lane >> 4) * 8;
    const int oa = n0 + nw + nl;
    const int ob = oa + 16;
    const float bva = bias[oa];
    const float bvb = bias[ob];
#pragma unroll
    for (int r = 0; r < 8; ++r) {
        const int s0 = m0 + mw + mbase + r;
        const int s1 = s0 + 16;
        out[(size_t)s0 * O_DIM + oa] = c00[r] + bva;
        out[(size_t)s0 * O_DIM + ob] = c01[r] + bvb;
        out[(size_t)s1 * O_DIM + oa] = c10[r] + bva;
        out[(size_t)s1 * O_DIM + ob] = c11[r] + bvb;
    }
}

// ---------------------------------------------------------------------------
// Host entry
// ---------------------------------------------------------------------------
extern "C" void kernel_launch(void* const* d_in, const int* in_sizes, int n_in,
                              void* d_out, int out_size, void* d_ws, size_t ws_size,
                              hipStream_t stream) {
    const float* x         = (const float*)d_in[0];
    const float* weight    = (const float*)d_in[1];
    const float* bias      = (const float*)d_in[2];
    const float* factor    = (const float*)d_in[3];
    const float* col_scale = (const float*)d_in[4];
    const float* col_bias  = (const float*)d_in[5];
    const float* Amat      = (const float*)d_in[6];
    const float* Bmat      = (const float*)d_in[7];
    float* out             = (float*)d_out;

    // workspace layout: xh (4MB) | wq (32MB) | wqt (32MB)
    char* ws = (char*)d_ws;
    _Float16* xh  = (_Float16*)ws;
    _Float16* wq  = (_Float16*)(ws + (size_t)S_DIM * I_DIM * 2);
    _Float16* wqt = (_Float16*)(ws + (size_t)S_DIM * I_DIM * 2 + (size_t)O_DIM * I_DIM * 2);

    cvt_x_kernel<<<(S_DIM * I_DIM) / (256 * 8), 256, 0, stream>>>(x, xh);

    quant_kernel<<<dim3(NGROUP, O_DIM), GS, 0, stream>>>(weight, factor, col_scale,
                                                         col_bias, Amat, Bmat, wq);

    trans_kernel<<<dim3(O_DIM / 64, I_DIM / 64), 256, 0, stream>>>(wq, wqt);

    gemm_kernel<<<dim3(S_DIM / 64, O_DIM / 128), 256, 0, stream>>>(xh, wqt, bias, out);
}
